// SpectralMoEDictionaryCrossAttention_2774548873639
// MI455X (gfx1250) — matrix operand order repaired and checked by reading, not compiled, not run on hardware
//
#include <hip/hip_runtime.h>
#include <hip/hip_bf16.h>
#include <stdint.h>

typedef unsigned short u16;
typedef __bf16 bf16_t;
typedef bf16_t bf16x16 __attribute__((ext_vector_type(16)));
typedef float f32x8 __attribute__((ext_vector_type(8)));

// ---------- scalar helpers ----------
__device__ __forceinline__ float bf2f(u16 h) {
    unsigned int u = ((unsigned int)h) << 16;
    return __builtin_bit_cast(float, u);
}
__device__ __forceinline__ u16 f2bf(float f) {
    unsigned int u = __builtin_bit_cast(unsigned int, f);
    unsigned int r = u + 0x7FFFu + ((u >> 16) & 1u);   // round-to-nearest-even
    return (u16)(r >> 16);
}
__device__ __forceinline__ float geluf(float x) {
    return 0.5f * x * (1.0f + erff(x * 0.7071067811865475f));
}
__device__ __forceinline__ float wsum(float x) {
    for (int o = 16; o > 0; o >>= 1) x += __shfl_xor(x, o, 32);
    return x;
}
__device__ __forceinline__ float wmaxr(float x) {
    for (int o = 16; o > 0; o >>= 1) x = fmaxf(x, __shfl_xor(x, o, 32));
    return x;
}

// ---------- fp32 -> bf16 convert ----------
__global__ __launch_bounds__(256) void k_cvt(const float* __restrict__ s, u16* __restrict__ d, int n) {
    int i = blockIdx.x * 256 + threadIdx.x;
    if (i < n) d[i] = f2bf(s[i]);
}

// ---------- NCHW f32 -> NHWC bf16 ----------
__global__ __launch_bounds__(256) void k_in(const float* __restrict__ x, u16* __restrict__ o,
                                            int B, int H, int W) {
    long long idx = (long long)blockIdx.x * 256 + threadIdx.x;
    long long tot = (long long)B * H * W * 128;
    if (idx >= tot) return;
    int c = (int)(idx & 127);
    long long t = idx >> 7;
    int w = (int)(t % W);
    int h = (int)((t / W) % H);
    int b = (int)(t / ((long long)W * H));
    o[idx] = f2bf(x[(((long long)b * 128 + c) * H + h) * W + w]);
}

// ---------- NHWC bf16 -> NCHW f32 (final output) ----------
__global__ __launch_bounds__(256) void k_outT(const u16* __restrict__ in, float* __restrict__ o,
                                              int B, int H, int W) {
    long long idx = (long long)blockIdx.x * 256 + threadIdx.x;
    long long tot = (long long)B * H * W * 128;
    if (idx >= tot) return;
    int w = (int)(idx % W);
    int h = (int)((idx / W) % H);
    int c = (int)((idx / ((long long)W * H)) % 128);
    int b = (int)(idx / ((long long)W * H * 128));
    o[idx] = bf2f(in[(((long long)b * H + h) * W + w) * 128 + c]);
}

// ---------- token LayerNorm over C=128 (bf16 or f32 source) ----------
__global__ __launch_bounds__(256) void k_ln(const u16* __restrict__ inb, const float* __restrict__ inf,
                                            const float* __restrict__ g, const float* __restrict__ bb,
                                            u16* __restrict__ out, int M) {
    int t = blockIdx.x * 8 + (threadIdx.x >> 5);
    int lane = threadIdx.x & 31;
    if (t >= M) return;
    size_t base = (size_t)t * 128 + lane * 4;
    float v[4];
    if (inf) { for (int r = 0; r < 4; ++r) v[r] = inf[base + r]; }
    else     { for (int r = 0; r < 4; ++r) v[r] = bf2f(inb[base + r]); }
    float s = v[0] + v[1] + v[2] + v[3];
    float s2 = v[0]*v[0] + v[1]*v[1] + v[2]*v[2] + v[3]*v[3];
    s = wsum(s); s2 = wsum(s2);
    float mean = s * (1.0f / 128.0f);
    float var = s2 * (1.0f / 128.0f) - mean * mean;
    float rs = rsqrtf(var + 1e-5f);
    int c0 = lane * 4;
    for (int r = 0; r < 4; ++r)
        out[base + r] = f2bf((v[r] - mean) * rs * g[c0 + r] + bb[c0 + r]);
}

// ---------- WMMA GEMM: C[M,N] = A[M,K] * Wt[N,K]^T + bias, bf16 in/out, f32 accum ----------
__global__ __launch_bounds__(256) void k_gemm(const u16* __restrict__ A, int lda,
                                              const u16* __restrict__ Wt,
                                              const float* __restrict__ bias,
                                              u16* __restrict__ Cc, int ldc,
                                              int M, int N, int K, int act) {
    __shared__ u16 As[64][40];    // 64x32 tile, +8 pad
    __shared__ u16 Bs[128][40];   // 128x32 tile (stored [n][k]), +8 pad
    const int tid = threadIdx.x;
    const int lane = tid & 31;
    const int wave = tid >> 5;
    const int wm = wave >> 2;     // 0..1
    const int wn = wave & 3;      // 0..3
    const int m0 = blockIdx.y * 64;
    const int n0 = blockIdx.x * 128;

    const int ar = tid >> 2;             // 0..63
    const int ac = (tid & 3) * 8;        // 0,8,16,24
    const int br = tid >> 1;             // 0..127
    const int bc = (tid & 1) * 16;       // 0,16

    const int frow = lane & 15;
    const int fkh = (lane >> 4) * 8;     // A half-K base
    const int fkb = (lane >> 4) * 8;     // B half-K base

    f32x8 acc[2][2];
    for (int i = 0; i < 2; ++i)
        for (int j = 0; j < 2; ++j)
            for (int r = 0; r < 8; ++r) acc[i][j][r] = 0.0f;

    for (int k0 = 0; k0 < K; k0 += 32) {
        uint4 av = {0, 0, 0, 0};
        if (m0 + ar < M) av = *(const uint4*)(A + (size_t)(m0 + ar) * lda + k0 + ac);
        uint4 bv0 = {0, 0, 0, 0}, bv1 = {0, 0, 0, 0};
        if (n0 + br < N) {
            const u16* wp = Wt + (size_t)(n0 + br) * K + k0 + bc;
            bv0 = *(const uint4*)(wp);
            bv1 = *(const uint4*)(wp + 8);
        }
        __syncthreads();
        *(uint4*)&As[ar][ac] = av;
        *(uint4*)&Bs[br][bc] = bv0;
        *(uint4*)&Bs[br][bc + 8] = bv1;
        __syncthreads();

        if (k0 + 32 < K) {  // emits global_prefetch_b8 for next K tile
            if (m0 + ar < M) __builtin_prefetch(A + (size_t)(m0 + ar) * lda + k0 + 32 + ac, 0, 0);
            if (n0 + br < N) __builtin_prefetch(Wt + (size_t)(n0 + br) * K + k0 + 32 + bc, 0, 0);
        }

        bf16x16 afrag[2], bfrag[2];
        for (int i = 0; i < 2; ++i) {
            const u16* ap = &As[wm * 32 + i * 16 + frow][0];
            ((uint4*)&afrag[i])[0] = *(const uint4*)(ap + fkh);        // K = half*8 .. +7
            ((uint4*)&afrag[i])[1] = *(const uint4*)(ap + 16 + fkh);   // K = 16+half*8 ..
        }
        for (int j = 0; j < 2; ++j) {
            const u16* bp = &Bs[wn * 32 + j * 16 + frow][0];
            ((uint4*)&bfrag[j])[0] = *(const uint4*)(bp + fkb);        // K = half*8 .. +7
            ((uint4*)&bfrag[j])[1] = *(const uint4*)(bp + 16 + fkb);   // K = 16+half*8 ..
        }
        for (int i = 0; i < 2; ++i)
            for (int j = 0; j < 2; ++j)
                acc[i][j] = __builtin_amdgcn_wmma_f32_16x16x32_bf16(
                    false, afrag[i], false, bfrag[j], (short)0, acc[i][j], false, false);
    }

    for (int i = 0; i < 2; ++i)
        for (int j = 0; j < 2; ++j) {
            int nb = n0 + wn * 32 + j * 16 + (lane & 15);
            int mb = m0 + wm * 32 + i * 16 + (lane >> 4) * 8;
            if (nb >= N) continue;
            float bz = bias ? bias[nb] : 0.0f;
            for (int r = 0; r < 8; ++r) {
                int m = mb + r;
                if (m >= M) break;
                float v = acc[i][j][r] + bz;
                if (act == 1) v = geluf(v);
                else if (act == 2) v = fmaxf(v, 0.0f);
                Cc[(size_t)m * ldc + nb] = f2bf(v);
            }
        }
}

// ---------- elementwise GELU with strides ----------
__global__ __launch_bounds__(256) void k_gelu(const u16* __restrict__ in, int ldin,
                                              u16* __restrict__ out, int ldout, int M) {
    long long idx = (long long)blockIdx.x * 256 + threadIdx.x;
    if (idx >= (long long)M * 128) return;
    int c = (int)(idx & 127);
    long long t = idx >> 7;
    out[t * ldout + c] = f2bf(geluf(bf2f(in[t * ldin + c])));
}

// ---------- depthwise 3x3, NHWC, pad 1, optional GELU ----------
__global__ __launch_bounds__(256) void k_dw3x3(const u16* __restrict__ in, int ldin,
                                               const float* __restrict__ w, const float* __restrict__ bias,
                                               u16* __restrict__ out, int ldout,
                                               int B, int H, int W, int C, int act) {
    long long idx = (long long)blockIdx.x * 256 + threadIdx.x;
    long long tot = (long long)B * H * W * C;
    if (idx >= tot) return;
    int c = (int)(idx % C);
    long long t = idx / C;
    int ww = (int)(t % W);
    int h = (int)((t / W) % H);
    int b = (int)(t / ((long long)W * H));
    float acc = bias[c];
    for (int ky = 0; ky < 3; ++ky) {
        int hh = h + ky - 1;
        if (hh < 0 || hh >= H) continue;
        for (int kx = 0; kx < 3; ++kx) {
            int wx = ww + kx - 1;
            if (wx < 0 || wx >= W) continue;
            acc += bf2f(in[(((long long)b * H + hh) * W + wx) * ldin + c]) * w[c * 9 + ky * 3 + kx];
        }
    }
    if (act) acc = geluf(acc);
    out[t * ldout + c] = f2bf(acc);
}

// ---------- channel mean/max pool ----------
__global__ __launch_bounds__(256) void k_pool(const u16* __restrict__ s, float* __restrict__ pool, int S) {
    int t = blockIdx.x * 8 + (threadIdx.x >> 5);
    int lane = threadIdx.x & 31;
    if (t >= S) return;
    float sm = 0.0f, mx = -3.4e38f;
    for (int r = 0; r < 4; ++r) {
        float v = bf2f(s[(size_t)t * 128 + lane + 32 * r]);
        sm += v; mx = fmaxf(mx, v);
    }
    sm = wsum(sm); mx = wmaxr(mx);
    if (lane == 0) { pool[2 * t] = sm * (1.0f / 128.0f); pool[2 * t + 1] = mx; }
}

// ---------- 7x7 2->1 conv + sigmoid (spatial attention) ----------
__global__ __launch_bounds__(256) void k_sa(const float* __restrict__ pool, const float* __restrict__ w,
                                            float* __restrict__ att, int B, int H, int W) {
    int t = blockIdx.x * 256 + threadIdx.x;
    if (t >= B * H * W) return;
    int ww = t % W;
    int h = (t / W) % H;
    int b = t / (W * H);
    float acc = 0.0f;
    for (int ky = 0; ky < 7; ++ky) {
        int hh = h + ky - 3;
        if (hh < 0 || hh >= H) continue;
        for (int kx = 0; kx < 7; ++kx) {
            int wx = ww + kx - 3;
            if (wx < 0 || wx >= W) continue;
            int tt = (b * H + hh) * W + wx;
            acc += pool[2 * tt] * w[ky * 7 + kx] + pool[2 * tt + 1] * w[49 + ky * 7 + kx];
        }
    }
    att[t] = 1.0f / (1.0f + __expf(-acc));
}

// ---------- x = s_out*att + x*res ----------
__global__ __launch_bounds__(256) void k_apply(const u16* __restrict__ s, const float* __restrict__ att,
                                               const float* __restrict__ res, u16* __restrict__ x, int S) {
    long long idx = (long long)blockIdx.x * 256 + threadIdx.x;
    if (idx >= (long long)S * 128) return;
    int c = (int)(idx & 127);
    long long t = idx >> 7;
    x[idx] = f2bf(bf2f(s[idx]) * att[t] + bf2f(x[idx]) * res[c]);
}

// ---------- x = a + x*res ----------
__global__ __launch_bounds__(256) void k_axpy(const u16* __restrict__ a, const float* __restrict__ res,
                                              u16* __restrict__ x, int S) {
    long long idx = (long long)blockIdx.x * 256 + threadIdx.x;
    if (idx >= (long long)S * 128) return;
    int c = (int)(idx & 127);
    x[idx] = f2bf(bf2f(a[idx]) + bf2f(x[idx]) * res[c]);
}

// ---------- Haar DWT (NHWC) ----------
__global__ __launch_bounds__(256) void k_dwt(const u16* __restrict__ X, u16* __restrict__ LL,
                                             u16* __restrict__ HF, int B, int Hh, int Wh) {
    long long idx = (long long)blockIdx.x * 256 + threadIdx.x;
    long long tot = (long long)B * Hh * Wh * 128;
    if (idx >= tot) return;
    int c = (int)(idx & 127);
    long long t = idx >> 7;
    int j = (int)(t % Wh);
    int i = (int)((t / Wh) % Hh);
    int b = (int)(t / ((long long)Wh * Hh));
    int H = Hh * 2, W = Wh * 2;
    size_t base = (((size_t)b * H + 2 * i) * W + 2 * j) * 128 + c;
    float a = bf2f(X[base]);
    float bb = bf2f(X[base + 128]);
    float cc = bf2f(X[base + (size_t)W * 128]);
    float dd = bf2f(X[base + (size_t)(W + 1) * 128]);
    float ll = 0.5f * (a + bb + cc + dd);
    float lh = 0.5f * (-a - bb + cc + dd);
    float hl = 0.5f * (-a + bb - cc + dd);
    float hh = 0.5f * (a - bb - cc + dd);
    LL[t * 128 + c] = f2bf(ll);
    size_t hb = t * 384 + (size_t)c * 3;
    HF[hb] = f2bf(lh); HF[hb + 1] = f2bf(hl); HF[hb + 2] = f2bf(hh);
}

// ---------- Haar IDWT (NHWC) ----------
__global__ __launch_bounds__(256) void k_idwt(const u16* __restrict__ LL, const u16* __restrict__ HF,
                                              u16* __restrict__ X, int B, int Hh, int Wh) {
    long long idx = (long long)blockIdx.x * 256 + threadIdx.x;
    long long tot = (long long)B * Hh * Wh * 128;
    if (idx >= tot) return;
    int c = (int)(idx & 127);
    long long t = idx >> 7;
    int j = (int)(t % Wh);
    int i = (int)((t / Wh) % Hh);
    int b = (int)(t / ((long long)Wh * Hh));
    int H = Hh * 2, W = Wh * 2;
    float ll = bf2f(LL[t * 128 + c]);
    size_t hb = t * 384 + (size_t)c * 3;
    float lh = bf2f(HF[hb]), hl = bf2f(HF[hb + 1]), hh = bf2f(HF[hb + 2]);
    size_t base = (((size_t)b * H + 2 * i) * W + 2 * j) * 128 + c;
    X[base]                          = f2bf(ll - lh - hl + hh);
    X[base + 128]                    = f2bf(ll - lh + hl - hh);
    X[base + (size_t)W * 128]        = f2bf(ll + lh - hl - hh);
    X[base + (size_t)(W + 1) * 128]  = f2bf(ll + lh + hl + hh);
}

// ---------- router softmax over 4 ----------
__global__ __launch_bounds__(256) void k_sm4(const u16* __restrict__ rl, float* __restrict__ rw, int S) {
    int t = blockIdx.x * 256 + threadIdx.x;
    if (t >= S) return;
    float v[4];
    float mx = -3.4e38f;
    for (int e = 0; e < 4; ++e) { v[e] = bf2f(rl[t * 4 + e]); mx = fmaxf(mx, v[e]); }
    float sm = 0.0f;
    for (int e = 0; e < 4; ++e) { v[e] = __expf(v[e] - mx); sm += v[e]; }
    float inv = 1.0f / sm;
    for (int e = 0; e < 4; ++e) rw[t * 4 + e] = v[e] * inv;
}

// ---------- dictionary cross-attention (wave per token, 4 heads, head dim 32) ----------
template <int NPER>  // entries / 32 : 2 (low, N=64) or 8 (high, N=256)
__global__ __launch_bounds__(256) void k_attn(const u16* __restrict__ Q, const u16* __restrict__ Kd,
                                              const u16* __restrict__ Vd, const float* __restrict__ scale,
                                              const float* __restrict__ rw, u16* __restrict__ Out, int Sq) {
    int t = blockIdx.x * 8 + (threadIdx.x >> 5);
    int lane = threadIdx.x & 31;
    if (t >= Sq) return;
    const int ne = NPER * 8;  // entries per expert (N/4)
    for (int e = 0; e < 4; ++e) {
        float q = bf2f(Q[(size_t)t * 128 + e * 32 + lane]);
        float sc = scale[e];
        float sims[NPER];
        for (int ci = 0; ci < NPER; ++ci) {
            int n = ci * 32 + lane;
            const u16* kr = Kd + (size_t)n * 128 + e * 32;
            u16 kk[32];
            *(uint4*)&kk[0]  = *(const uint4*)(kr);
            *(uint4*)&kk[8]  = *(const uint4*)(kr + 8);
            *(uint4*)&kk[16] = *(const uint4*)(kr + 16);
            *(uint4*)&kk[24] = *(const uint4*)(kr + 24);
            float d = 0.0f;
#pragma unroll
            for (int c = 0; c < 32; ++c) d += bf2f(kk[c]) * __shfl(q, c, 32);
            sims[ci] = d * sc;
        }
        float mx = -3.4e38f;
        for (int ci = 0; ci < NPER; ++ci) mx = fmaxf(mx, sims[ci]);
        mx = wmaxr(mx);
        float sm = 0.0f;
        for (int ci = 0; ci < NPER; ++ci) { sims[ci] = __expf(sims[ci] - mx); sm += sims[ci]; }
        sm = wsum(sm);
        float inv = 1.0f / sm;
        for (int ci = 0; ci < NPER; ++ci) {
            sims[ci] *= inv;
            if (rw) sims[ci] *= rw[(size_t)t * 4 + ((ci * 32 + lane) / ne)];
        }
        float acc = 0.0f;
        for (int ci = 0; ci < NPER; ++ci) {
            float ps = sims[ci];
#pragma unroll
            for (int k = 0; k < 32; ++k) {
                float p = __shfl(ps, k, 32);
                acc += p * bf2f(Vd[(size_t)(ci * 32 + k) * 128 + e * 32 + lane]);
            }
        }
        Out[(size_t)t * 128 + e * 32 + lane] = f2bf(acc);
    }
}

// ---------- MLP gate: g = gelu(dw3x3(a)) * v, a/v = split(fc1out) ----------
__global__ __launch_bounds__(256) void k_gate(const u16* __restrict__ h, const float* __restrict__ w,
                                              const float* __restrict__ bias, u16* __restrict__ G,
                                              int B, int H, int W) {
    long long idx = (long long)blockIdx.x * 256 + threadIdx.x;
    long long tot = (long long)B * H * W * 256;
    if (idx >= tot) return;
    int c = (int)(idx % 256);
    long long t = idx / 256;
    int ww = (int)(t % W);
    int hh0 = (int)((t / W) % H);
    int b = (int)(t / ((long long)W * H));
    float acc = bias[c];
    for (int ky = 0; ky < 3; ++ky) {
        int hy = hh0 + ky - 1;
        if (hy < 0 || hy >= H) continue;
        for (int kx = 0; kx < 3; ++kx) {
            int wx = ww + kx - 1;
            if (wx < 0 || wx >= W) continue;
            acc += bf2f(h[(((long long)b * H + hy) * W + wx) * 512 + c]) * w[c * 9 + ky * 3 + kx];
        }
    }
    G[t * 256 + c] = f2bf(geluf(acc) * bf2f(h[t * 512 + 256 + c]));
}

// =====================================================================

extern "C" void kernel_launch(void* const* d_in, const int* in_sizes, int n_in,
                              void* d_out, int out_size, void* d_ws, size_t ws_size,
                              hipStream_t stream) {
    (void)in_sizes; (void)n_in; (void)out_size; (void)ws_size;
    const int Bn = 4, Hn = 192, Wn = 192;
    const int S = Bn * Hn * Wn;          // 147456
    const int Hh = Hn / 2, Wh = Wn / 2;
    const int Sh = Bn * Hh * Wh;         // 36864

    size_t off = 0;
    auto alloc = [&](size_t bytes) -> void* {
        off = (off + 255) & ~(size_t)255;
        void* p = (char*)d_ws + off;
        off += bytes;
        return p;
    };
    auto F = [&](int i) -> const float* { return (const float*)d_in[i]; };

    u16* XB   = (u16*)alloc((size_t)S * 128 * 2);
    u16* TA   = (u16*)alloc((size_t)S * 128 * 2);
    u16* TB   = (u16*)alloc((size_t)S * 128 * 2);
    u16* CAT  = (u16*)alloc((size_t)S * 512 * 2);
    u16* G    = (u16*)alloc((size_t)S * 256 * 2);
    float* POOL = (float*)alloc((size_t)S * 2 * 4);
    float* ATT  = (float*)alloc((size_t)S * 4);
    u16* LLb  = (u16*)alloc((size_t)Sh * 128 * 2);
    u16* HFb  = (u16*)alloc((size_t)Sh * 384 * 2);
    u16* HFE  = (u16*)alloc((size_t)Sh * 128 * 2);
    u16* QT   = (u16*)alloc((size_t)Sh * 128 * 2);
    u16* LLO  = (u16*)alloc((size_t)Sh * 128 * 2);
    u16* HO   = (u16*)alloc((size_t)Sh * 128 * 2);
    u16* HFO  = (u16*)alloc((size_t)Sh * 384 * 2);
    u16* R1   = (u16*)alloc((size_t)Sh * 32 * 2);
    u16* RL   = (u16*)alloc((size_t)Sh * 4 * 2);
    float* RW = (float*)alloc((size_t)Sh * 4 * 4);
    u16* DN   = (u16*)alloc(64 * 128 * 2);
    u16* EN   = (u16*)alloc(256 * 128 * 2);
    u16* KL   = (u16*)alloc(64 * 128 * 2);
    u16* KH   = (u16*)alloc(256 * 128 * 2);

    auto cvt = [&](int idx, size_t n) -> const u16* {
        u16* dst = (u16*)alloc(n * 2);
        k_cvt<<<dim3((unsigned)((n + 255) / 256)), 256, 0, stream>>>((const float*)d_in[idx], dst, (int)n);
        return dst;
    };
    // param index map (setup_inputs dict order, x at 0)
    const u16* w_xt = cvt(1, 128 * 128);
    const u16* w_s  = cvt(5, 128 * 128);
    const u16* w_din[3];
    const u16* w_dout[3];
    for (int i = 0; i < 3; ++i) {
        w_din[i]  = cvt(7 + 6 * i, 128 * 128);
        w_dout[i] = cvt(11 + 6 * i, 128 * 128);
    }
    const u16* w_proj = cvt(25, 128 * 512);
    const u16* w_ql   = cvt(33, 128 * 128);
    const u16* w_kl   = cvt(35, 128 * 128);
    const u16* w_r1   = cvt(41, 32 * 128);
    const u16* w_r2   = cvt(43, 4 * 32);
    const u16* w_hin  = cvt(45, 128 * 384);
    const u16* w_hout = cvt(47, 384 * 128);
    const u16* w_qh   = cvt(51, 128 * 128);
    const u16* w_kh   = cvt(53, 128 * 128);
    const u16* w_lin  = cvt(58, 128 * 128);
    const u16* w_fc1  = cvt(62, 512 * 128);
    const u16* w_fc2  = cvt(66, 128 * 256);
    const u16* w_out  = cvt(68, 128 * 128);

    auto gemm = [&](const u16* A, int lda, const u16* Wt, const float* bias,
                    u16* C, int ldc, int M, int N, int K, int act) {
        dim3 g((N + 127) / 128, (M + 63) / 64);
        k_gemm<<<g, 256, 0, stream>>>(A, lda, Wt, bias, C, ldc, M, N, K, act);
    };
    auto LN = [&](const u16* inb, const float* inf, const float* g_, const float* b_, u16* o, int M) {
        k_ln<<<dim3((M + 7) / 8), 256, 0, stream>>>(inb, inf, g_, b_, o, M);
    };
    const unsigned gS = (unsigned)(((long long)S * 128 + 255) / 256);
    const unsigned gSh = (unsigned)(((long long)Sh * 128 + 255) / 256);

    // input transpose + x_trans
    k_in<<<gS, 256, 0, stream>>>((const float*)d_in[0], TA, Bn, Hn, Wn);
    gemm(TA, 128, w_xt, F(2), XB, 128, S, 128, 128, 0);

    // MSA
    LN(XB, nullptr, F(3), F(4), TA, S);
    gemm(TA, 128, w_s, F(6), CAT, 512, S, 128, 128, 0);
    for (int i = 0; i < 3; ++i) {
        k_gelu<<<gS, 256, 0, stream>>>(CAT + i * 128, 512, TB, 128, S);
        gemm(TB, 128, w_din[i], F(8 + 6 * i), TA, 128, S, 128, 128, 1);
        k_dw3x3<<<gS, 256, 0, stream>>>(TA, 128, F(9 + 6 * i), F(10 + 6 * i), TB, 128, Bn, Hn, Wn, 128, 1);
        gemm(TB, 128, w_dout[i], F(12 + 6 * i), CAT + (i + 1) * 128, 512, S, 128, 128, 0);
    }
    gemm(CAT, 512, w_proj, F(26), TB, 128, S, 128, 512, 0);
    k_pool<<<dim3((S + 7) / 8), 256, 0, stream>>>(TB, POOL, S);
    k_sa<<<dim3((S + 255) / 256), 256, 0, stream>>>(POOL, F(27), ATT, Bn, Hn, Wn);
    k_apply<<<gS, 256, 0, stream>>>(TB, ATT, F(28), XB, S);   // XB = msa + x*res1 (XB is shortcut)

    // DWT
    k_dwt<<<gSh, 256, 0, stream>>>(XB, LLb, HFb, Bn, Hh, Wh);

    // low attention
    LN(LLb, nullptr, F(31), F(32), TA, Sh);
    gemm(TA, 128, w_ql, F(34), QT, 128, Sh, 128, 128, 0);
    LN(nullptr, F(39), F(37), F(38), DN, 64);
    gemm(DN, 128, w_kl, F(36), KL, 128, 64, 128, 128, 0);
    k_attn<2><<<dim3((Sh + 7) / 8), 256, 0, stream>>>(QT, KL, DN, F(40), nullptr, LLO, Sh);

    // high path
    gemm(HFb, 384, w_hin, F(46), HFE, 128, Sh, 128, 384, 0);
    gemm(HFE, 128, w_r1, F(42), R1, 32, Sh, 32, 128, 2);
    gemm(R1, 32, w_r2, F(44), RL, 4, Sh, 4, 32, 0);
    k_sm4<<<dim3((Sh + 255) / 256), 256, 0, stream>>>(RL, RW, Sh);
    LN(HFE, nullptr, F(49), F(50), TA, Sh);
    gemm(TA, 128, w_qh, F(52), QT, 128, Sh, 128, 128, 0);
    LN(nullptr, F(57), F(55), F(56), EN, 256);
    gemm(EN, 128, w_kh, F(54), KH, 128, 256, 128, 128, 0);
    k_attn<8><<<dim3((Sh + 7) / 8), 256, 0, stream>>>(QT, KH, EN, F(40), RW, HO, Sh);
    gemm(HO, 128, w_hout, F(48), HFO, 384, Sh, 384, 128, 0);

    // IDWT + linear + residual
    k_idwt<<<gSh, 256, 0, stream>>>(LLO, HFO, TA, Bn, Hh, Wh);
    gemm(TA, 128, w_lin, F(59), TB, 128, S, 128, 128, 0);
    k_axpy<<<gS, 256, 0, stream>>>(TB, F(29), XB, S);

    // MLP
    LN(XB, nullptr, F(60), F(61), TA, S);
    gemm(TA, 128, w_fc1, F(63), CAT, 512, S, 512, 128, 0);
    k_gate<<<dim3((unsigned)(((long long)S * 256 + 255) / 256)), 256, 0, stream>>>(CAT, F(64), F(65), G, Bn, Hn, Wn);
    gemm(G, 256, w_fc2, F(67), TB, 128, S, 128, 256, 0);
    k_axpy<<<gS, 256, 0, stream>>>(TB, F(30), XB, S);

    // output projection + NCHW f32
    gemm(XB, 128, w_out, F(69), TB, 128, S, 128, 128, 0);
    k_outT<<<gS, 256, 0, stream>>>(TB, (float*)d_out, Bn, Hn, Wn);
}
